// GraphSpatiotemporalEncoder_57191784513609
// MI455X (gfx1250) — matrix operand order repaired and checked by reading, not compiled, not run on hardware
//
#include <hip/hip_runtime.h>
#include <hip/hip_bf16.h>
#include <cstdint>

#define H_GAT   4
#define NEG_SL  0.2f
#define LN_EPS  1e-5f
#define GHID    64
#define HC      256          // H_GAT * GHID
#define DTOK    512
#define TRH     8

typedef __attribute__((ext_vector_type(16))) __bf16 v16bf;
typedef __attribute__((ext_vector_type(8)))  float  v8f;

static inline int cdiv(int a, int b) { return (a + b - 1) / b; }

// ---------- monotone float<->uint encoding for atomicMax on floats ----------
__device__ __forceinline__ unsigned f2ord(float f) {
    unsigned u = __float_as_uint(f);
    return (u & 0x80000000u) ? ~u : (u | 0x80000000u);
}
__device__ __forceinline__ float ord2f(unsigned u) {
    unsigned v = (u & 0x80000000u) ? (u & 0x7FFFFFFFu) : ~u;
    return __uint_as_float(v);
}

__device__ __forceinline__ v16bf cvt16(const float4& f0, const float4& f1,
                                       const float4& f2, const float4& f3) {
    float f[16] = {f0.x,f0.y,f0.z,f0.w, f1.x,f1.y,f1.z,f1.w,
                   f2.x,f2.y,f2.z,f2.w, f3.x,f3.y,f3.z,f3.w};
    v16bf v;
#pragma unroll
    for (int e = 0; e < 16; ++e) v[e] = (__bf16)f[e];
    return v;
}

// =====================================================================
// WMMA bf16 GEMM:  C[M,N] = act(A[M,K] @ W[N,K]^T + bias)
// A row-major f32, W row-major f32 [N,K].
// One wave computes a 16x64 C tile. All 20 b128 loads for a K-step are
// issued up front (one clause), then 4 independent convert+WMMA chains
// drain them with descending s_wait_loadcnt values.
// Requires M%16==0, N%64==0, K%32==0 (true for every call site here).
//
// Fragment layouts (16-bit data, wave32; A/B K order chosen here, C/D per ISA §7.12.2):
//  A 16x32: e -> K = 16*half + e (16 contiguous K per lane, matching B)
//  B 32x16: e -> K = 16*half + e (16 contiguous K per lane)
//  C/D 16x16 f32: VGPR r -> M = r + 8*half, N = lane&15
// =====================================================================
__global__ void gemm_bf16_wmma(const float* __restrict__ A,
                               const float* __restrict__ W,
                               const float* __restrict__ bias,
                               float* __restrict__ C,
                               int M, int N, int K, int act)
{
    const int lane = threadIdx.x & 31;
    const int wave = threadIdx.x >> 5;
    const int wpb  = blockDim.x >> 5;
    const int tile = blockIdx.x * wpb + wave;
    const int tilesN = N >> 6;               // 64-wide C tiles
    const int tm = tile / tilesN;
    const int tn = tile - tm * tilesN;
    if (tm * 16 >= M) return;

    const int half = lane >> 4;
    const int l15  = lane & 15;
    const float* arow = A + (size_t)(tm * 16 + l15) * K;
    const float* wrow[4];
#pragma unroll
    for (int s = 0; s < 4; ++s)
        wrow[s] = W + (size_t)(tn * 64 + 16 * s + l15) * K + 16 * half;

    v8f acc[4] = {{}, {}, {}, {}};
    for (int k0 = 0; k0 < K; k0 += 32) {
        // ---- issue ALL loads for this K-step up front (single clause) ----
        float4 ar0 = *reinterpret_cast<const float4*>(arow + k0 + 16 * half);
        float4 ar1 = *reinterpret_cast<const float4*>(arow + k0 + 16 * half + 4);
        float4 ar2 = *reinterpret_cast<const float4*>(arow + k0 + 16 * half + 8);
        float4 ar3 = *reinterpret_cast<const float4*>(arow + k0 + 16 * half + 12);
        float4 br[4][4];
#pragma unroll
        for (int s = 0; s < 4; ++s) {
            const float* p = wrow[s] + k0;
#pragma unroll
            for (int q = 0; q < 4; ++q)
                br[s][q] = *reinterpret_cast<const float4*>(p + 4 * q);
        }
        if (k0 + 32 < K)                       // global_prefetch_b8 next A slab
            __builtin_prefetch(arow + k0 + 32, 0, 3);

        // ---- convert + 4 independent WMMAs ----
        v16bf av = cvt16(ar0, ar1, ar2, ar3);
#pragma unroll
        for (int s = 0; s < 4; ++s) {
            v16bf bv = cvt16(br[s][0], br[s][1], br[s][2], br[s][3]);
            acc[s] = __builtin_amdgcn_wmma_f32_16x16x32_bf16(
                         false, av, false, bv, (short)0, acc[s], false, false);
        }
    }

#pragma unroll
    for (int s = 0; s < 4; ++s) {
#pragma unroll
        for (int r = 0; r < 8; ++r) {
            int m = tm * 16 + r + 8 * half;
            int n = tn * 64 + 16 * s + l15;
            float v = acc[s][r];
            if (bias) v += bias[n];
            if (act)  v = fmaxf(v, 0.f);
            C[(size_t)m * N + n] = v;
        }
    }
}

// =====================================================================
// Self-loop edge-attr (fill_value='mean'): per-target mean of incoming attrs
// =====================================================================
__global__ void sl_accum(const float* __restrict__ eattr, const int* __restrict__ dst,
                         float* __restrict__ sl, float* __restrict__ cnt,
                         int T, int E, int N)
{
    int e = blockIdx.x * blockDim.x + threadIdx.x;
    if (e >= E) return;
    int d = dst[e];
    atomicAdd(&cnt[d], 1.0f);
    for (int t = 0; t < T; ++t) {
        atomicAdd(&sl[((size_t)t * N + d) * 2 + 0], eattr[((size_t)t * E + e) * 2 + 0]);
        atomicAdd(&sl[((size_t)t * N + d) * 2 + 1], eattr[((size_t)t * E + e) * 2 + 1]);
    }
}

__global__ void sl_div(float* __restrict__ sl, const float* __restrict__ cnt, int T, int N)
{
    int i = blockIdx.x * blockDim.x + threadIdx.x;
    if (i >= T * N) return;
    int n = i % N;
    float c = fmaxf(cnt[n], 1.0f);
    sl[(size_t)i * 2 + 0] /= c;
    sl[(size_t)i * 2 + 1] /= c;
}

// =====================================================================
// GATv2 edge passes
// =====================================================================
__global__ void gat_logit_max(const float* __restrict__ xl, const float* __restrict__ xr,
                              const float* __restrict__ eattr, const float* __restrict__ sl,
                              const int* __restrict__ src, const int* __restrict__ dst,
                              const float* __restrict__ we, const float* __restrict__ att,
                              float* __restrict__ logit, unsigned* __restrict__ smax,
                              int T, int E, int N, int E2)
{
    int idx = blockIdx.x * blockDim.x + threadIdx.x;
    if (idx >= T * E2 * H_GAT) return;
    int hh = idx % H_GAT;
    int te = idx / H_GAT;
    int e  = te % E2;
    int t  = te / E2;
    int s_, d_; float a0, a1;
    if (e < E) {
        s_ = src[e]; d_ = dst[e];
        a0 = eattr[((size_t)t * E + e) * 2 + 0];
        a1 = eattr[((size_t)t * E + e) * 2 + 1];
    } else {
        int n = e - E; s_ = n; d_ = n;
        a0 = sl[((size_t)t * N + n) * 2 + 0];
        a1 = sl[((size_t)t * N + n) * 2 + 1];
    }
    const float* xls  = xl + ((size_t)t * N + s_) * HC + hh * GHID;
    const float* xrd  = xr + ((size_t)t * N + d_) * HC + hh * GHID;
    const float* wrow = we + (size_t)hh * GHID * 2;
    const float* arow = att + hh * GHID;
    float acc = 0.f;
#pragma unroll 4
    for (int c = 0; c < GHID; ++c) {
        float m = xls[c] + xrd[c] + a0 * wrow[c * 2] + a1 * wrow[c * 2 + 1];
        m = (m > 0.f) ? m : NEG_SL * m;     // leaky_relu
        acc += m * arow[c];
    }
    logit[idx] = acc;
    atomicMax(&smax[((size_t)t * N + d_) * H_GAT + hh], f2ord(acc));
}

__global__ void gat_exp_sum(float* __restrict__ logit, const unsigned* __restrict__ smax,
                            float* __restrict__ ssum, const int* __restrict__ dst,
                            int T, int E, int N, int E2)
{
    int idx = blockIdx.x * blockDim.x + threadIdx.x;
    if (idx >= T * E2 * H_GAT) return;
    int hh = idx % H_GAT;
    int te = idx / H_GAT;
    int e  = te % E2;
    int t  = te / E2;
    int d_ = (e < E) ? dst[e] : (e - E);
    size_t nd = ((size_t)t * N + d_) * H_GAT + hh;
    float ex = __expf(logit[idx] - ord2f(smax[nd]));
    logit[idx] = ex;
    atomicAdd(&ssum[nd], ex);
}

__global__ void gat_aggregate(const float* __restrict__ logit, const float* __restrict__ ssum,
                              const float* __restrict__ xl,
                              const int* __restrict__ src, const int* __restrict__ dst,
                              float* __restrict__ gout, int T, int E, int N, int E2)
{
    int idx = blockIdx.x * blockDim.x + threadIdx.x;
    if (idx >= T * E2 * H_GAT) return;
    int hh = idx % H_GAT;
    int te = idx / H_GAT;
    int e  = te % E2;
    int t  = te / E2;
    int s_ = (e < E) ? src[e] : (e - E);
    int d_ = (e < E) ? dst[e] : (e - E);
    size_t nd = ((size_t)t * N + d_) * H_GAT + hh;
    float alpha = logit[idx] / ssum[nd];
    const float* xls = xl + ((size_t)t * N + s_) * HC + hh * GHID;
    float* o = gout + ((size_t)t * N + d_) * HC + hh * GHID;
#pragma unroll 4
    for (int c = 0; c < GHID; ++c)
        atomicAdd(&o[c], alpha * xls[c]);
}

// head-mean + bias + relu + residual + LayerNorm(64), one wave32 per row
__global__ void gat_post(const float* __restrict__ gout, const float* __restrict__ gb,
                         const float* __restrict__ lw, const float* __restrict__ lb,
                         float* __restrict__ h, int rows)
{
    int lane = threadIdx.x & 31;
    int wave = threadIdx.x >> 5;
    int row  = blockIdx.x * (blockDim.x >> 5) + wave;
    if (row >= rows) return;
    float v[2];
#pragma unroll
    for (int j = 0; j < 2; ++j) {
        int c = lane + 32 * j;
        const float* g = gout + (size_t)row * HC;
        float s = 0.f;
#pragma unroll
        for (int hh = 0; hh < H_GAT; ++hh) s += g[hh * GHID + c];
        s = s * (1.0f / H_GAT) + gb[c];
        v[j] = fmaxf(s, 0.f) + h[(size_t)row * GHID + c];
    }
    float sum = v[0] + v[1];
#pragma unroll
    for (int o = 16; o >= 1; o >>= 1) sum += __shfl_xor(sum, o, 32);
    float mu = sum * (1.0f / GHID);
    float d0 = v[0] - mu, d1 = v[1] - mu;
    float sq = d0 * d0 + d1 * d1;
#pragma unroll
    for (int o = 16; o >= 1; o >>= 1) sq += __shfl_xor(sq, o, 32);
    float inv = rsqrtf(sq * (1.0f / GHID) + LN_EPS);
#pragma unroll
    for (int j = 0; j < 2; ++j) {
        int c = lane + 32 * j;
        h[(size_t)row * GHID + c] = (v[j] - mu) * inv * lw[c] + lb[c];
    }
}

// =====================================================================
// Graph pooling (mean + max per graph)
// =====================================================================
__global__ void pool_cnt(const int* __restrict__ batch, float* __restrict__ cntB, int N)
{
    int n = blockIdx.x * blockDim.x + threadIdx.x;
    if (n >= N) return;
    atomicAdd(&cntB[batch[n]], 1.0f);
}

__global__ void pool_acc(const float* __restrict__ h, const int* __restrict__ batch,
                         float* __restrict__ psum, unsigned* __restrict__ pmax,
                         int T, int N, int Bn)
{
    int idx = blockIdx.x * blockDim.x + threadIdx.x;
    if (idx >= T * N * GHID) return;
    int c = idx % GHID;
    int n = (idx / GHID) % N;
    int t = idx / (GHID * N);
    float v = h[idx];
    int b = batch[n];
    size_t o = ((size_t)t * Bn + b) * GHID + c;
    atomicAdd(&psum[o], v);
    atomicMax(&pmax[o], f2ord(v));
}

__global__ void pool_finish(const float* __restrict__ psum, const unsigned* __restrict__ pmax,
                            const float* __restrict__ cntB, float* __restrict__ pooled,
                            int T, int Bn)
{
    int idx = blockIdx.x * blockDim.x + threadIdx.x;
    if (idx >= T * Bn * 2 * GHID) return;
    int j  = idx % (2 * GHID);
    int b  = (idx / (2 * GHID)) % Bn;
    int t  = idx / (2 * GHID * Bn);
    size_t base = ((size_t)t * Bn + b) * GHID;
    pooled[idx] = (j < GHID) ? psum[base + j] / cntB[b]
                             : ord2f(pmax[base + (j - GHID)]);
}

// [T,B,D] -> [B,T,D]
__global__ void transpose_tok(const float* __restrict__ tok, float* __restrict__ y,
                              int T, int Bn, int D)
{
    int idx = blockIdx.x * blockDim.x + threadIdx.x;
    if (idx >= Bn * T * D) return;
    int d = idx % D;
    int t = (idx / D) % T;
    int b = idx / (D * T);
    y[idx] = tok[((size_t)t * Bn + b) * D + d];
}

__global__ void motion_kernel(const float* __restrict__ y, float* __restrict__ mo,
                              int T, int Bn, int D)
{
    int idx = blockIdx.x * blockDim.x + threadIdx.x;
    if (idx >= Bn * (T - 1) * 256) return;
    int d = idx % 256;
    int t = (idx / 256) % (T - 1);
    int b = idx / (256 * (T - 1));
    mo[idx] = y[((size_t)b * T + t + 1) * D + d] - y[((size_t)b * T + t) * D + d];
}

// =====================================================================
// MHA core: one block per (batch, head); T==8, dh==64, 64 threads
// =====================================================================
__global__ void mha_kernel(const float* __restrict__ qkv, float* __restrict__ out,
                           int Bn, int T, int D)
{
    int b  = blockIdx.x / TRH;
    int hh = blockIdx.x % TRH;
    int dh = D / TRH;                  // 64
    int tid = threadIdx.x;             // 64 threads
    __shared__ float sc[8][8];
    __shared__ float at[8][8];
    int i = tid >> 3, j = tid & 7;
    const float* qb = qkv + (size_t)(b * T) * 3 * D + hh * dh;
    const float* kb = qb + D;
    const float* vb = qb + 2 * D;

    float dot = 0.f;
    for (int d = 0; d < dh; ++d)
        dot += qb[(size_t)i * 3 * D + d] * kb[(size_t)j * 3 * D + d];
    sc[i][j] = dot * rsqrtf((float)dh);
    __syncthreads();

    float mx = sc[i][0];
#pragma unroll
    for (int jj = 1; jj < 8; ++jj) mx = fmaxf(mx, sc[i][jj]);
    float ex = __expf(sc[i][j] - mx);
    at[i][j] = ex;
    __syncthreads();
    float den = 0.f;
#pragma unroll
    for (int jj = 0; jj < 8; ++jj) den += at[i][jj];
    __syncthreads();
    at[i][j] = ex / den;
    __syncthreads();

    int d = tid;                       // 0..63
    for (int ii = 0; ii < 8; ++ii) {
        float o = 0.f;
#pragma unroll
        for (int jj = 0; jj < 8; ++jj) o += at[ii][jj] * vb[(size_t)jj * 3 * D + d];
        out[(size_t)(b * T + ii) * D + hh * dh + d] = o;
    }
}

// y = LayerNorm(y + xin); one block (256 threads) per row
__global__ void ln_residual(const float* __restrict__ xin, float* __restrict__ y,
                            const float* __restrict__ w, const float* __restrict__ bb,
                            int Dims)
{
    int row = blockIdx.x;
    int tid = threadIdx.x;
    __shared__ float red[256];
    float v[8];
    int cnt = 0;
    float lsum = 0.f;
    for (int c = tid; c < Dims; c += 256) {
        float t = xin[(size_t)row * Dims + c] + y[(size_t)row * Dims + c];
        v[cnt++] = t;
        lsum += t;
    }
    red[tid] = lsum; __syncthreads();
    for (int o = 128; o > 0; o >>= 1) { if (tid < o) red[tid] += red[tid + o]; __syncthreads(); }
    float mu = red[0] / Dims;
    __syncthreads();
    float lsq = 0.f;
    for (int k = 0; k < cnt; ++k) { float d = v[k] - mu; lsq += d * d; }
    red[tid] = lsq; __syncthreads();
    for (int o = 128; o > 0; o >>= 1) { if (tid < o) red[tid] += red[tid + o]; __syncthreads(); }
    float inv = rsqrtf(red[0] / Dims + LN_EPS);
    __syncthreads();
    int k = 0;
    for (int c = tid; c < Dims; c += 256) {
        y[(size_t)row * Dims + c] = (v[k] - mu) * inv * w[c] + bb[c];
        ++k;
    }
}

// =====================================================================
// Host driver
// =====================================================================
static inline void gemm(const float* A, const float* W, const float* bias, float* C,
                        int M, int N, int K, int act, hipStream_t s)
{
    int tiles = (M / 16) * (N / 64);             // 16x64 C tile per wave
    int blocks = cdiv(tiles, 8);                 // 8 waves (wave32) per 256-thread block
    gemm_bf16_wmma<<<blocks, 256, 0, s>>>(A, W, bias, C, M, N, K, act);
}

extern "C" void kernel_launch(void* const* d_in, const int* in_sizes, int n_in,
                              void* d_out, int out_size, void* d_ws, size_t ws_size,
                              hipStream_t stream)
{
    const float* x       = (const float*)d_in[0];
    const int*   ei      = (const int*)d_in[1];
    const float* eattr   = (const float*)d_in[2];
    const int*   batch   = (const int*)d_in[3];
    const float* proj_w  = (const float*)d_in[4];
    const float* proj_b  = (const float*)d_in[5];
    const float* gat_wl  = (const float*)d_in[6];
    const float* gat_wr  = (const float*)d_in[7];
    const float* gat_we  = (const float*)d_in[8];
    const float* gat_att = (const float*)d_in[9];
    const float* gat_b   = (const float*)d_in[10];
    const float* ln_w    = (const float*)d_in[11];
    const float* ln_b    = (const float*)d_in[12];
    const float* n2t_w   = (const float*)d_in[13];
    const float* n2t_b   = (const float*)d_in[14];
    const float* tr_wqkv = (const float*)d_in[15];
    const float* tr_bqkv = (const float*)d_in[16];
    const float* tr_wo   = (const float*)d_in[17];
    const float* tr_bo   = (const float*)d_in[18];
    const float* tr_w1   = (const float*)d_in[19];
    const float* tr_b1   = (const float*)d_in[20];
    const float* tr_w2   = (const float*)d_in[21];
    const float* tr_b2   = (const float*)d_in[22];
    const float* tr_ln1w = (const float*)d_in[23];
    const float* tr_ln1b = (const float*)d_in[24];
    const float* tr_ln2w = (const float*)d_in[25];
    const float* tr_ln2b = (const float*)d_in[26];

    const int E  = in_sizes[1] / 2;
    const int N  = in_sizes[3];
    const int T  = in_sizes[0] / (N * 128);
    const int Bn = 8;
    const int E2 = E + N;
    const int D  = DTOK;
    const int* src = ei;
    const int* dst = ei + E;

    // ---------------- workspace carve (bump allocator) ----------------
    char* wp = (char*)d_ws;
    auto alloc = [&](size_t bytes) -> void* {
        char* p = wp;
        wp += (bytes + 255) & ~(size_t)255;
        return (void*)p;
    };
    float*    cnt   = (float*)alloc(sizeof(float) * N);
    float*    cntB  = (float*)alloc(sizeof(float) * Bn);
    float*    sl    = (float*)alloc(sizeof(float) * (size_t)T * N * 2);
    float*    h     = (float*)alloc(sizeof(float) * (size_t)T * N * GHID);
    float*    xl    = (float*)alloc(sizeof(float) * (size_t)T * N * HC);
    float*    xr    = (float*)alloc(sizeof(float) * (size_t)T * N * HC);
    float*    gout  = xr;   // alias: xr only needed in logit pass, gout only after it
    float*    logit = (float*)alloc(sizeof(float) * (size_t)T * E2 * H_GAT);
    unsigned* smax  = (unsigned*)alloc(sizeof(unsigned) * (size_t)T * N * H_GAT);
    float*    ssum  = (float*)alloc(sizeof(float) * (size_t)T * N * H_GAT);
    float*    psum  = (float*)alloc(sizeof(float) * (size_t)T * Bn * GHID);
    unsigned* pmax  = (unsigned*)alloc(sizeof(unsigned) * (size_t)T * Bn * GHID);
    float*    pooled= (float*)alloc(sizeof(float) * (size_t)T * Bn * 2 * GHID);
    float*    tok   = (float*)alloc(sizeof(float) * (size_t)T * Bn * D);
    float*    qkvb  = (float*)alloc(sizeof(float) * (size_t)Bn * T * 3 * D);
    float*    attn  = (float*)alloc(sizeof(float) * (size_t)Bn * T * D);
    float*    tmpd  = (float*)alloc(sizeof(float) * (size_t)Bn * T * D);
    float*    ff1   = (float*)alloc(sizeof(float) * (size_t)Bn * T * 4 * D);
    (void)ws_size; (void)n_in; (void)out_size;

    float* y      = (float*)d_out;                       // [B*T, D]
    float* motion = (float*)d_out + (size_t)Bn * T * D;  // [B, T-1, 256]

    const int TH = 256;
    const int NE = T * E2 * H_GAT;

    // ---------------- self-loop edge attrs ----------------
    hipMemsetAsync(cnt, 0, sizeof(float) * N, stream);
    hipMemsetAsync(sl, 0, sizeof(float) * (size_t)T * N * 2, stream);
    sl_accum<<<cdiv(E, TH), TH, 0, stream>>>(eattr, dst, sl, cnt, T, E, N);
    sl_div<<<cdiv(T * N, TH), TH, 0, stream>>>(sl, cnt, T, N);

    // ---------------- input projection (WMMA) ----------------
    gemm(x, proj_w, proj_b, h, T * N, GHID, 128, 0, stream);

    // ---------------- 3x GATv2 + LN ----------------
    for (int i = 0; i < 3; ++i) {
        gemm(h, gat_wl + (size_t)i * HC * GHID, nullptr, xl, T * N, HC, GHID, 0, stream);
        gemm(h, gat_wr + (size_t)i * HC * GHID, nullptr, xr, T * N, HC, GHID, 0, stream);
        hipMemsetAsync(smax, 0, sizeof(unsigned) * (size_t)T * N * H_GAT, stream);
        hipMemsetAsync(ssum, 0, sizeof(float) * (size_t)T * N * H_GAT, stream);
        gat_logit_max<<<cdiv(NE, TH), TH, 0, stream>>>(
            xl, xr, eattr, sl, src, dst,
            gat_we + (size_t)i * HC * 2, gat_att + (size_t)i * H_GAT * GHID,
            logit, smax, T, E, N, E2);
        gat_exp_sum<<<cdiv(NE, TH), TH, 0, stream>>>(logit, smax, ssum, dst, T, E, N, E2);
        hipMemsetAsync(gout, 0, sizeof(float) * (size_t)T * N * HC, stream);
        gat_aggregate<<<cdiv(NE, TH), TH, 0, stream>>>(logit, ssum, xl, src, dst, gout, T, E, N, E2);
        gat_post<<<cdiv(T * N, 8), 256, 0, stream>>>(
            gout, gat_b + (size_t)i * GHID, ln_w + (size_t)i * GHID, ln_b + (size_t)i * GHID,
            h, T * N);
    }

    // ---------------- pooling + node->token (WMMA) ----------------
    hipMemsetAsync(cntB, 0, sizeof(float) * Bn, stream);
    hipMemsetAsync(psum, 0, sizeof(float) * (size_t)T * Bn * GHID, stream);
    hipMemsetAsync(pmax, 0, sizeof(unsigned) * (size_t)T * Bn * GHID, stream);
    pool_cnt<<<cdiv(N, TH), TH, 0, stream>>>(batch, cntB, N);
    pool_acc<<<cdiv(T * N * GHID, TH), TH, 0, stream>>>(h, batch, psum, pmax, T, N, Bn);
    pool_finish<<<cdiv(T * Bn * 2 * GHID, TH), TH, 0, stream>>>(psum, pmax, cntB, pooled, T, Bn);
    gemm(pooled, n2t_w, n2t_b, tok, T * Bn, D, 2 * GHID, 0, stream);
    transpose_tok<<<cdiv(Bn * T * D, TH), TH, 0, stream>>>(tok, y, T, Bn, D);
    motion_kernel<<<cdiv(Bn * (T - 1) * 256, TH), TH, 0, stream>>>(y, motion, T, Bn, D);

    // ---------------- 2x post-norm transformer layers ----------------
    for (int i = 0; i < 2; ++i) {
        gemm(y, tr_wqkv + (size_t)i * 3 * D * D, tr_bqkv + (size_t)i * 3 * D,
             qkvb, Bn * T, 3 * D, D, 0, stream);
        mha_kernel<<<Bn * TRH, 64, 0, stream>>>(qkvb, attn, Bn, T, D);
        gemm(attn, tr_wo + (size_t)i * D * D, tr_bo + (size_t)i * D,
             tmpd, Bn * T, D, D, 0, stream);
        ln_residual<<<Bn * T, 256, 0, stream>>>(tmpd, y, tr_ln1w + (size_t)i * D,
                                                tr_ln1b + (size_t)i * D, D);
        gemm(y, tr_w1 + (size_t)i * 4 * D * D, tr_b1 + (size_t)i * 4 * D,
             ff1, Bn * T, 4 * D, D, 1, stream);
        gemm(ff1, tr_w2 + (size_t)i * D * 4 * D, tr_b2 + (size_t)i * D,
             tmpd, Bn * T, D, 4 * D, 0, stream);
        ln_residual<<<Bn * T, 256, 0, stream>>>(tmpd, y, tr_ln2w + (size_t)i * D,
                                                tr_ln2b + (size_t)i * D, D);
    }
}
